// LeanAlbertAttention_15375982920090
// MI455X (gfx1250) — matrix operand order repaired and checked by reading, not compiled, not run on hardware
//
#include <hip/hip_runtime.h>

// ---------------------------------------------------------------------------
// LeanAlbertAttention on MI455X (gfx1250, wave32, WMMA + TDM).
// All matmuls on v_wmma_f32_16x16x32_bf16 (fp32 accum). Weight panels are
// streamed into double-buffered LDS by the Tensor Data Mover (TDM), overlapped
// with WMMA compute; synchronization via TENSORcnt + block barriers.
// Fragment layouts (from CDNA5 ISA 7.12.2):
//   A (16x32): lane m=lane%16, two contiguous 8-elem runs at k=8*(lane/16)
//              and k=16+8*(lane/16)               -> row-major [M][K]
//   B (32x16): lane n=lane%16, two contiguous 8-elem runs at k=8*(lane/16)
//              and k=16+8*(lane/16) -> row-major B^T [N][K]
//   C (16x16): lane n=lane%16, element i -> m=i+8*(lane/16)
// ---------------------------------------------------------------------------

typedef __attribute__((ext_vector_type(16))) __bf16 v16bf;
typedef __attribute__((ext_vector_type(8)))  __bf16 bf16x8;
typedef __attribute__((ext_vector_type(8)))  float  v8f;
typedef __attribute__((ext_vector_type(4)))  unsigned int u32x4;
typedef __attribute__((ext_vector_type(8)))  int    i32x8;
typedef __attribute__((ext_vector_type(4)))  int    i32x4;

#define HIDDEN  1024
#define NHEADS  16
#define HDIM    64
#define SEQ     2048
#define BATCH   2
#define ROWS    (BATCH * SEQ)        // 4096 tokens
#define QKV_N   (3 * HIDDEN)         // 3072
#define KCH     128                  // K-chunk staged per TDM transfer

// workspace offsets (bytes); total 48 MiB
#define OFF_XLN    (0ull)                       // bf16 [4096][1024]   8 MiB
#define OFF_WQKVT  (8ull  * 1024 * 1024)        // bf16 [3072][1024]   6 MiB
#define OFF_WOUTT  (14ull * 1024 * 1024)        // bf16 [1024][1024]   2 MiB
#define OFF_Q      (16ull * 1024 * 1024)        // bf16 [b][h][s][d]   8 MiB
#define OFF_K      (24ull * 1024 * 1024)        // bf16 [b][h][s][d]   8 MiB
#define OFF_VT     (32ull * 1024 * 1024)        // bf16 [b][h][d][s]   8 MiB
#define OFF_CTX    (40ull * 1024 * 1024)        // bf16 [4096][1024]   8 MiB

// Build a 16-element bf16 fragment from two contiguous 8-element (16B) runs.
static __device__ inline v16bf frag_from(const __bf16* lo, const __bf16* hi) {
  bf16x8 l = *(const bf16x8*)lo;
  bf16x8 h = *(const bf16x8*)hi;
  v16bf f;
#pragma unroll
  for (int i = 0; i < 8; ++i) { f[i] = l[i]; f[i + 8] = h[i]; }
  return f;
}

static __device__ inline v16bf load_fragA(const __bf16* base, int ld,
                                          int kbase, int mlane, int half) {
  const __bf16* r = base + (size_t)mlane * ld + kbase + 8 * half;
  return frag_from(r, r + 16);
}

static __device__ inline v16bf load_fragB(const __bf16* baseT, int ld,
                                          int kbase, int nlane, int half) {
  const __bf16* r = baseT + (size_t)nlane * ld + kbase + 8 * half;
  return frag_from(r, r + 16);
}

static __device__ inline v8f wmma_bf16(v16bf a, v16bf b, v8f c) {
  return __builtin_amdgcn_wmma_f32_16x16x32_bf16(false, a, false, b,
                                                 (short)0, c, false, false);
}

// ---------------------------------------------------------------------------
// TDM: async 2D tile load Global -> LDS. D# per CDNA5 ISA ch.8:
//  group0: [1:0]=count=1, [63:32]=lds_addr, [120:64]=global_addr, type=2
//  group1: data_size=2B, tensor_dim0/1, tile_dim0/1, tensor_dim0_stride
// Tracked by TENSORcnt; EXEC-ignored (issue from one wave, fence with
// s_wait_tensorcnt + workgroup barrier). 6-arg builtin form (this toolchain).
// ---------------------------------------------------------------------------
static __device__ inline void tdm_load_2d(unsigned lds_byte_off,
                                          const __bf16* gaddr,
                                          unsigned tensor_d0,  // row len (elems)
                                          unsigned tensor_d1,  // #rows
                                          unsigned tile_d0,    // tile cols
                                          unsigned tile_d1,    // tile rows
                                          unsigned stride0) {  // row stride
  unsigned long long ga = (unsigned long long)(uintptr_t)gaddr;
  u32x4 g0;
  g0[0] = 1u;                                        // count=1, user D#
  g0[1] = lds_byte_off;
  g0[2] = (unsigned)(ga & 0xffffffffu);
  g0[3] = (unsigned)((ga >> 32) & 0x01ffffffu) | (2u << 30);  // type=2
  i32x8 g1;
  g1[0] = (int)(1u << 16);                           // data_size = 2 bytes
  g1[1] = (int)((tensor_d0 & 0xffffu) << 16);
  g1[2] = (int)(((tensor_d0 >> 16) & 0xffffu) | ((tensor_d1 & 0xffffu) << 16));
  g1[3] = (int)(((tensor_d1 >> 16) & 0xffffu) | ((tile_d0 & 0xffffu) << 16));
  g1[4] = (int)(tile_d1 & 0xffffu);                  // tile_dim1, tile_dim2=0
  g1[5] = (int)stride0;                              // stride0 low 32
  g1[6] = 0;                                         // stride0 hi / stride1 lo
  g1[7] = 0;
  i32x4 z4 = {0, 0, 0, 0};
  i32x8 z8 = {0, 0, 0, 0, 0, 0, 0, 0};
  __builtin_amdgcn_tensor_load_to_lds(g0, g1, z4, z4, z8, 0);
}

// ---------------------------------------------------------------------------
// 1) LayerNorm fp32 -> bf16. One wave32 per token row (32 lanes x 32 elems).
// ---------------------------------------------------------------------------
__global__ void ln_bf16_kernel(const float* __restrict__ x,
                               const float* __restrict__ gamma,
                               const float* __restrict__ beta,
                               __bf16* __restrict__ out) {
  int row  = blockIdx.x * (blockDim.x >> 5) + (threadIdx.x >> 5);
  int lane = threadIdx.x & 31;
  const float* xr = x + (size_t)row * HIDDEN;

  float v[32];
  float s = 0.f;
#pragma unroll
  for (int i = 0; i < 32; ++i) { v[i] = xr[lane + 32 * i]; s += v[i]; }
#pragma unroll
  for (int o = 16; o > 0; o >>= 1) s += __shfl_xor(s, o, 32);
  float mu = s * (1.0f / HIDDEN);

  float var = 0.f;
#pragma unroll
  for (int i = 0; i < 32; ++i) { float d = v[i] - mu; var += d * d; }
#pragma unroll
  for (int o = 16; o > 0; o >>= 1) var += __shfl_xor(var, o, 32);
  float rstd = rsqrtf(var * (1.0f / HIDDEN) + 1e-12f);

  __bf16* orow = out + (size_t)row * HIDDEN;
#pragma unroll
  for (int i = 0; i < 32; ++i) {
    int c = lane + 32 * i;
    orow[c] = (__bf16)((v[i] - mu) * rstd * gamma[c] + beta[c]);
  }
}

// ---------------------------------------------------------------------------
// 2) Weight transpose + fp32 -> bf16:  Wt[N][K] = W[K][N]  (one-time cost)
// ---------------------------------------------------------------------------
__global__ void transpose_bf16_kernel(const float* __restrict__ W,
                                      __bf16* __restrict__ Wt,
                                      int R, int C) {
  int c = blockIdx.x * blockDim.x + threadIdx.x;
  int r = blockIdx.y;
  if (c < C) Wt[(size_t)c * R + r] = (__bf16)W[(size_t)r * C + c];
}

// ---------------------------------------------------------------------------
// Shared GEMM mainloop: block = 8 waves, block tile 256(M) x 64(N),
// wave tile 32 x 64 (8 accumulators). B panel [64][KCH] double-buffered in
// LDS, streamed by TDM (wave 0 issues; transfer for chunk c+1 overlaps WMMA
// on chunk c). Per K-step: 2 A frags (global) + 4 B frags (LDS) -> 8 WMMA.
// ---------------------------------------------------------------------------
#define GEMM_MAINLOOP(ACC, ABASE, BSH, WT_PANEL, TENSOR_D1)                    \
  do {                                                                         \
    unsigned lds0 = (unsigned)(uintptr_t)(&BSH[0][0][0]);                      \
    unsigned lds1 = (unsigned)(uintptr_t)(&BSH[1][0][0]);                      \
    if (wave == 0)                                                             \
      tdm_load_2d(lds0, (WT_PANEL), HIDDEN, (TENSOR_D1), KCH, 64, HIDDEN);     \
    if (wave == 0) __builtin_amdgcn_s_wait_tensorcnt(0);                       \
    __syncthreads();                                                           \
    for (int c = 0; c < HIDDEN / KCH; ++c) {                                   \
      if (wave == 0 && c + 1 < HIDDEN / KCH)                                   \
        tdm_load_2d((c & 1) ? lds0 : lds1, (WT_PANEL) + (c + 1) * KCH,         \
                    HIDDEN, (TENSOR_D1), KCH, 64, HIDDEN);                     \
      const __bf16* bp = &BSH[c & 1][0][0];                                    \
      for (int ks = 0; ks < KCH; ks += 32) {                                   \
        int kg = c * KCH + ks;                                                 \
        v16bf a0 = load_fragA((ABASE), HIDDEN, kg, mlane, half);               \
        v16bf a1 = load_fragA((ABASE) + 16 * HIDDEN, HIDDEN, kg, mlane, half); \
        v16bf b0 = load_fragB(bp +  0 * KCH, KCH, ks, mlane, half);            \
        v16bf b1 = load_fragB(bp + 16 * KCH, KCH, ks, mlane, half);            \
        v16bf b2 = load_fragB(bp + 32 * KCH, KCH, ks, mlane, half);            \
        v16bf b3 = load_fragB(bp + 48 * KCH, KCH, ks, mlane, half);            \
        ACC[0] = wmma_bf16(a0, b0, ACC[0]);                                    \
        ACC[1] = wmma_bf16(a0, b1, ACC[1]);                                    \
        ACC[2] = wmma_bf16(a0, b2, ACC[2]);                                    \
        ACC[3] = wmma_bf16(a0, b3, ACC[3]);                                    \
        ACC[4] = wmma_bf16(a1, b0, ACC[4]);                                    \
        ACC[5] = wmma_bf16(a1, b1, ACC[5]);                                    \
        ACC[6] = wmma_bf16(a1, b2, ACC[6]);                                    \
        ACC[7] = wmma_bf16(a1, b3, ACC[7]);                                    \
      }                                                                        \
      if (wave == 0) __builtin_amdgcn_s_wait_tensorcnt(0);                     \
      __syncthreads();                                                         \
    }                                                                          \
  } while (0)

// ---------------------------------------------------------------------------
// 3) QKV GEMM: [4096 x 1024] x [1024 x 3072] + bias.
//    Q scaled by 1/sqrt(64), written [b][h][s][d]; K [b][h][s][d];
//    V transposed [b][h][d][s].
// ---------------------------------------------------------------------------
__global__ void qkv_gemm_kernel(const __bf16* __restrict__ xln,
                                const __bf16* __restrict__ wT,   // [3072][1024]
                                const float*  __restrict__ bias, // [3072]
                                __bf16* __restrict__ Q,
                                __bf16* __restrict__ K,
                                __bf16* __restrict__ Vt) {
  __shared__ __bf16 Bsh[2][64][KCH];   // 32 KiB double-buffered weight panel

  int wave  = threadIdx.x >> 5;
  int lane  = threadIdx.x & 31;
  int mlane = lane & 15;
  int half  = lane >> 4;
  int m0 = blockIdx.x * 256 + wave * 32;
  int n0 = blockIdx.y * 64;

  const __bf16* A = xln + (size_t)m0 * HIDDEN;
  v8f acc[8] = {};

  GEMM_MAINLOOP(acc, A, Bsh, wT + (size_t)n0 * HIDDEN, QKV_N);

#pragma unroll
  for (int tm = 0; tm < 2; ++tm) {
#pragma unroll
    for (int t = 0; t < 4; ++t) {
#pragma unroll
      for (int i = 0; i < 8; ++i) {
        int m = m0 + tm * 16 + i + 8 * half;   // token
        int n = n0 + t * 16 + mlane;           // qkv column
        float val = acc[tm * 4 + t][i] + bias[n];
        int b   = m >> 11;
        int s   = m & (SEQ - 1);
        int sec = n >> 10;                     // 0=Q 1=K 2=V
        int nn  = n & (HIDDEN - 1);
        int h   = nn >> 6;
        int d   = nn & (HDIM - 1);
        size_t bh = (size_t)(b * NHEADS + h);
        if (sec == 0)
          Q[(bh * SEQ + s) * HDIM + d] = (__bf16)(val * 0.125f);
        else if (sec == 1)
          K[(bh * SEQ + s) * HDIM + d] = (__bf16)val;
        else
          Vt[(bh * HDIM + d) * SEQ + s] = (__bf16)val;
      }
    }
  }
}

// ---------------------------------------------------------------------------
// 4) Attention, flash-style. One wave per (b, h, 16-query tile).
//    Per 32-key step: 4 batched K-frag loads -> 4 WMMA (scores), online
//    softmax via half-wave __shfl_xor reductions, P restaged through LDS,
//    4 batched V-frag loads -> 4 WMMA (P @ V). Next tile prefetched.
// ---------------------------------------------------------------------------
__global__ void attention_kernel(const __bf16* __restrict__ Q,
                                 const __bf16* __restrict__ K,
                                 const __bf16* __restrict__ Vt,
                                 const float*  __restrict__ mask, // [B][S]
                                 __bf16* __restrict__ ctx) {      // [4096][1024]
  __shared__ __bf16 Pbuf[8][16][32];   // per-wave P tile (16 q x 32 k)

  int wave  = threadIdx.x >> 5;
  int lane  = threadIdx.x & 31;
  int mlane = lane & 15;
  int half  = lane >> 4;

  int gw = blockIdx.x * 8 + wave;      // 0..4095
  int qt = gw & 127;
  int bh = gw >> 7;
  int b  = bh >> 4;
  int h  = bh & 15;

  const __bf16* Qb = Q  + ((size_t)bh * SEQ + qt * 16) * HDIM;
  const __bf16* Kb = K  + (size_t)bh * SEQ * HDIM;
  const __bf16* Vb = Vt + (size_t)bh * HDIM * SEQ;
  const float*  mk = mask + (size_t)b * SEQ;

  v16bf qa[2];
#pragma unroll
  for (int j = 0; j < 2; ++j) qa[j] = load_fragA(Qb, HDIM, 32 * j, mlane, half);

  v8f   acc[4] = {};
  float mrow[8], lrow[8];
#pragma unroll
  for (int i = 0; i < 8; ++i) { mrow[i] = -3.0e38f; lrow[i] = 0.f; }

  for (int kt = 0; kt < SEQ; kt += 32) {
    // prefetch next key tile of K and V into cache while computing this one
    if (kt + 32 < SEQ) {
      __builtin_prefetch(Kb + (size_t)(kt + 32) * HDIM + lane * 4, 0, 3);
      __builtin_prefetch(Vb + (size_t)mlane * SEQ + kt + 32, 0, 3);
    }

    // ---- scores: batched fragment loads, then 4 WMMA ----
    v16bf kb0 = load_fragB(Kb + (size_t)kt * HDIM,        HDIM,  0, mlane, half);
    v16bf kb1 = load_fragB(Kb + (size_t)(kt + 16) * HDIM, HDIM,  0, mlane, half);
    v16bf kb2 = load_fragB(Kb + (size_t)kt * HDIM,        HDIM, 32, mlane, half);
    v16bf kb3 = load_fragB(Kb + (size_t)(kt + 16) * HDIM, HDIM, 32, mlane, half);
    v8f s0 = {}, s1 = {};
    s0 = wmma_bf16(qa[0], kb0, s0);
    s1 = wmma_bf16(qa[0], kb1, s1);
    s0 = wmma_bf16(qa[1], kb2, s0);
    s1 = wmma_bf16(qa[1], kb3, s1);

    float mk0 = mk[kt + mlane];
    float mk1 = mk[kt + 16 + mlane];

    // ---- online softmax ----
    float alpha[8];
#pragma unroll
    for (int i = 0; i < 8; ++i) {
      float a0 = s0[i] + mk0;
      float a1 = s1[i] + mk1;
      float mx = fmaxf(a0, a1);
#pragma unroll
      for (int o = 1; o < 16; o <<= 1) mx = fmaxf(mx, __shfl_xor(mx, o, 32));
      float mnew = fmaxf(mrow[i], mx);
      alpha[i] = __expf(mrow[i] - mnew);
      mrow[i]  = mnew;
      float p0 = __expf(a0 - mnew);
      float p1 = __expf(a1 - mnew);
      float rs = p0 + p1;
#pragma unroll
      for (int o = 1; o < 16; o <<= 1) rs += __shfl_xor(rs, o, 32);
      lrow[i] = lrow[i] * alpha[i] + rs;
      int r = i + 8 * half;
      Pbuf[wave][r][mlane]      = (__bf16)p0;
      Pbuf[wave][r][16 + mlane] = (__bf16)p1;
    }
#pragma unroll
    for (int t = 0; t < 4; ++t)
#pragma unroll
      for (int i = 0; i < 8; ++i) acc[t][i] *= alpha[i];

    // ---- P @ V : batched fragment loads, then 4 WMMA ----
    v16bf pa = load_fragA(&Pbuf[wave][0][0], 32, 0, mlane, half);
    v16bf vb0 = load_fragB(Vb +  0 * SEQ, SEQ, kt, mlane, half);
    v16bf vb1 = load_fragB(Vb + 16 * SEQ, SEQ, kt, mlane, half);
    v16bf vb2 = load_fragB(Vb + 32 * SEQ, SEQ, kt, mlane, half);
    v16bf vb3 = load_fragB(Vb + 48 * SEQ, SEQ, kt, mlane, half);
    acc[0] = wmma_bf16(pa, vb0, acc[0]);
    acc[1] = wmma_bf16(pa, vb1, acc[1]);
    acc[2] = wmma_bf16(pa, vb2, acc[2]);
    acc[3] = wmma_bf16(pa, vb3, acc[3]);
  }

  // ---- normalize and write ctx [token][h*64 + d] ----
#pragma unroll
  for (int t = 0; t < 4; ++t) {
#pragma unroll
    for (int i = 0; i < 8; ++i) {
      int q = qt * 16 + i + 8 * half;
      int d = t * 16 + mlane;
      float o = acc[t][i] / lrow[i];
      ctx[((size_t)b * SEQ + q) * HIDDEN + h * HDIM + d] = (__bf16)o;
    }
  }
}

// ---------------------------------------------------------------------------
// 5) Out projection + bias + residual: [4096 x 1024] x [1024 x 1024] -> fp32
// ---------------------------------------------------------------------------
__global__ void out_gemm_kernel(const __bf16* __restrict__ ctx,
                                const __bf16* __restrict__ wT,    // [1024][1024]
                                const float*  __restrict__ bias,  // [1024]
                                const float*  __restrict__ resid, // [4096][1024]
                                float* __restrict__ out) {
  __shared__ __bf16 Bsh[2][64][KCH];   // 32 KiB double-buffered weight panel

  int wave  = threadIdx.x >> 5;
  int lane  = threadIdx.x & 31;
  int mlane = lane & 15;
  int half  = lane >> 4;
  int m0 = blockIdx.x * 256 + wave * 32;
  int n0 = blockIdx.y * 64;

  const __bf16* A = ctx + (size_t)m0 * HIDDEN;
  v8f acc[8] = {};

  GEMM_MAINLOOP(acc, A, Bsh, wT + (size_t)n0 * HIDDEN, HIDDEN);

#pragma unroll
  for (int tm = 0; tm < 2; ++tm) {
#pragma unroll
    for (int t = 0; t < 4; ++t) {
#pragma unroll
      for (int i = 0; i < 8; ++i) {
        int m = m0 + tm * 16 + i + 8 * half;
        int n = n0 + t * 16 + mlane;
        out[(size_t)m * HIDDEN + n] =
            acc[tm * 4 + t][i] + bias[n] + resid[(size_t)m * HIDDEN + n];
      }
    }
  }
}

// ---------------------------------------------------------------------------
extern "C" void kernel_launch(void* const* d_in, const int* in_sizes, int n_in,
                              void* d_out, int out_size, void* d_ws,
                              size_t ws_size, hipStream_t stream) {
  const float* hidden = (const float*)d_in[0];
  const float* mask   = (const float*)d_in[1];
  const float* Wqkv   = (const float*)d_in[2];
  const float* bqkv   = (const float*)d_in[3];
  const float* Wout   = (const float*)d_in[4];
  const float* bout   = (const float*)d_in[5];
  const float* gamma  = (const float*)d_in[6];
  const float* beta   = (const float*)d_in[7];
  float* out = (float*)d_out;

  char* ws = (char*)d_ws;
  __bf16* xln   = (__bf16*)(ws + OFF_XLN);
  __bf16* wqkvT = (__bf16*)(ws + OFF_WQKVT);
  __bf16* woutT = (__bf16*)(ws + OFF_WOUTT);
  __bf16* Qb    = (__bf16*)(ws + OFF_Q);
  __bf16* Kb    = (__bf16*)(ws + OFF_K);
  __bf16* Vtb   = (__bf16*)(ws + OFF_VT);
  __bf16* ctxb  = (__bf16*)(ws + OFF_CTX);

  // 1) LayerNorm -> bf16 (4096 rows, 8 rows/block)
  ln_bf16_kernel<<<dim3(ROWS / 8), dim3(256), 0, stream>>>(hidden, gamma, beta,
                                                           xln);
  // 2) Weight transposes -> bf16 [N][K]
  transpose_bf16_kernel<<<dim3(QKV_N / 256, HIDDEN), dim3(256), 0, stream>>>(
      Wqkv, wqkvT, HIDDEN, QKV_N);
  transpose_bf16_kernel<<<dim3(HIDDEN / 256, HIDDEN), dim3(256), 0, stream>>>(
      Wout, woutT, HIDDEN, HIDDEN);
  // 3) QKV GEMM (M=4096, N=3072, K=1024), TDM-staged B panels
  qkv_gemm_kernel<<<dim3(ROWS / 256, QKV_N / 64), dim3(256), 0, stream>>>(
      xln, wqkvT, bqkv, Qb, Kb, Vtb);
  // 4) Attention (4096 wave-tiles, 8 waves/block)
  attention_kernel<<<dim3(512), dim3(256), 0, stream>>>(Qb, Kb, Vtb, mask,
                                                        ctxb);
  // 5) Out projection + bias + residual (M=4096, N=1024, K=1024)
  out_gemm_kernel<<<dim3(ROWS / 256, HIDDEN / 64), dim3(256), 0, stream>>>(
      ctxb, woutT, bout, hidden, out);
}
